// GATEncoder_49039936586203
// MI455X (gfx1250) — matrix operand order repaired and checked by reading, not compiled, not run on hardware
//
#include <hip/hip_runtime.h>
#include <cstdint>

// ---------------------------------------------------------------------------
// CDNA5 (gfx1250) GAT encoder.
//   GEMMs: v_wmma_f32_16x16x32_bf16, wave32; block tile 16x256, BK=64,
//          4 WMMAs per barrier pair; weights pre-packed as K-pair dwords;
//          tiles staged with GLOBAL_LOAD_ASYNC_TO_LDS (ASYNCcnt) when the
//          builtin exists, synchronous vector copies otherwise.
//   Edge softmax/aggregation: global f32 atomics (L2-resident working set).
// ---------------------------------------------------------------------------

typedef __attribute__((ext_vector_type(16))) __bf16 v16bf;
typedef __attribute__((ext_vector_type(8)))  float  v8f;
typedef int vi4 __attribute__((vector_size(16)));   // int __vector(4)

union AFrag { v16bf v; unsigned u[8]; };
union BFrag { v16bf v; unsigned u[8]; };

__device__ __forceinline__ unsigned short f2bf(float f) {
    unsigned u = __float_as_uint(f);
    unsigned r = ((u >> 16) & 1u) + 0x7FFFu;   // round-to-nearest-even
    return (unsigned short)((u + r) >> 16);
}

// monotone float<->uint key for atomic float max
__device__ __forceinline__ unsigned fkey(float f) {
    unsigned u = __float_as_uint(f);
    return (u & 0x80000000u) ? ~u : (u | 0x80000000u);
}
__device__ __forceinline__ float funkey(unsigned k) {
    unsigned u = (k & 0x80000000u) ? (k & 0x7FFFFFFFu) : ~k;
    return __uint_as_float(u);
}

// ---- global -> LDS tile staging (async when available) --------------------
#if __has_builtin(__builtin_amdgcn_global_load_async_to_lds_b128) && \
    __has_builtin(__builtin_amdgcn_global_load_async_to_lds_b32)  && \
    __has_builtin(__builtin_amdgcn_s_wait_asynccnt)
#define USE_ASYNC_COPY 1
__device__ __forceinline__ void cp_lds128(void* lds, const void* g) {
    __builtin_amdgcn_global_load_async_to_lds_b128(
        (__attribute__((address_space(1))) vi4*)g,
        (__attribute__((address_space(3))) vi4*)lds, 0, 0);
}
__device__ __forceinline__ void cp_lds32(void* lds, const void* g) {
    __builtin_amdgcn_global_load_async_to_lds_b32(
        (__attribute__((address_space(1))) int*)g,
        (__attribute__((address_space(3))) int*)lds, 0, 0);
}
__device__ __forceinline__ void cp_wait() { __builtin_amdgcn_s_wait_asynccnt(0); }
#else
#define USE_ASYNC_COPY 0
__device__ __forceinline__ void cp_lds128(void* lds, const void* g) {
    *(float4*)lds = *(const float4*)g;
}
__device__ __forceinline__ void cp_lds32(void* lds, const void* g) {
    *(unsigned*)lds = *(const unsigned*)g;
}
__device__ __forceinline__ void cp_wait() {}
#endif

// ---------------------------------------------------------------------------
// WMMA GEMM: C[M,N] = A[M,K] * B[K,N]; A bf16 row-major, Bp = packed K-pair
// dwords [(K/2) x N], C f32. M%16==0, N%256==0, K%64==0.
// 256 thr = 8 waves; wave w owns C columns [32w, 32w+32) as two 16x16 tiles.
// ---------------------------------------------------------------------------
__global__ __launch_bounds__(256) void wmma_gemm_bf16(
    const unsigned short* __restrict__ A,
    const unsigned* __restrict__ Bp,
    float* __restrict__ C, int M, int N, int K)
{
    __shared__ unsigned Apack[32][20];    // [K-pair][m]  (+4 pad: 16-bank skew)
    __shared__ unsigned Blds[32][260];    // [K-pair][n]  (+4 pad: 32-bank skew)

    const int tid  = threadIdx.x;
    const int lane = tid & 31;
    const int wave = tid >> 5;
    const int lm   = lane & 15;
    const int g    = lane >> 4;          // half-wave group
    const int m0   = blockIdx.y * 16;
    const int n0   = blockIdx.x * 256;
    const int Kd   = K >> 1;             // A row length in pair-dwords
    const unsigned* Adw = (const unsigned*)A;

    v8f acc0 = {}, acc1 = {};

    for (int k0 = 0; k0 < K; k0 += 64) {
        const int kp0 = k0 >> 1;
        // A tile: 32 K-pairs x 16 rows = 512 dwords (2 per thread)
        {
            int i0 = tid, i1 = tid + 256;
            int p0 = i0 >> 4, r0 = i0 & 15;
            int p1 = i1 >> 4, r1 = i1 & 15;
            cp_lds32(&Apack[p0][r0], &Adw[(size_t)(m0 + r0) * Kd + kp0 + p0]);
            cp_lds32(&Apack[p1][r1], &Adw[(size_t)(m0 + r1) * Kd + kp0 + p1]);
        }
        // B tile: 32 K-pairs x 256 cols = 2048 x b128 chunks (8 per thread)
        {
            int idx = tid;
#pragma unroll
            for (int i = 0; i < 8; ++i) {
                int p = idx >> 6, c = (idx & 63) * 4;
                cp_lds128(&Blds[p][c], &Bp[(size_t)(kp0 + p) * N + n0 + c]);
                idx += 256;
            }
        }
        cp_wait();
        __syncthreads();

        AFrag a0, a1;
        BFrag b00, b01, b10, b11;
        // A 16x32 bf16 frag: vgpr v -> pair v+8g
#pragma unroll
        for (int v = 0; v < 8; ++v) {
            int pa = v + 8 * g;
            a0.u[v] = Apack[pa][lm];           // K =  0..31 of this BK slab
            a1.u[v] = Apack[pa + 16][lm];      // K = 32..63
        }
        // B 32x16 bf16 frag: vgpr v -> pair v+8g, col = lane%16
        const int nb0 = wave * 32 + lm, nb1 = nb0 + 16;
#pragma unroll
        for (int v = 0; v < 8; ++v) {
            int pb = v + 8 * g;
            b00.u[v] = Blds[pb][nb0];
            b01.u[v] = Blds[pb][nb1];
            b10.u[v] = Blds[pb + 16][nb0];
            b11.u[v] = Blds[pb + 16][nb1];
        }
        acc0 = __builtin_amdgcn_wmma_f32_16x16x32_bf16(
                   false, a0.v, false, b00.v, (short)0, acc0, false, false);
        acc0 = __builtin_amdgcn_wmma_f32_16x16x32_bf16(
                   false, a1.v, false, b10.v, (short)0, acc0, false, false);
        acc1 = __builtin_amdgcn_wmma_f32_16x16x32_bf16(
                   false, a0.v, false, b01.v, (short)0, acc1, false, false);
        acc1 = __builtin_amdgcn_wmma_f32_16x16x32_bf16(
                   false, a1.v, false, b11.v, (short)0, acc1, false, false);
        __syncthreads();
    }

    // C/D layout: vgpr r -> M = r + 8g, N = lane%16
    float* c0 = C + (size_t)m0 * N + n0 + wave * 32 + lm;
#pragma unroll
    for (int r = 0; r < 8; ++r) {
        c0[(size_t)(r + 8 * g) * N]      = acc0[r];
        c0[(size_t)(r + 8 * g) * N + 16] = acc1[r];
    }
}

// ---------------------------------------------------------------------------
__global__ __launch_bounds__(256) void f32_to_bf16_k(
    const float* __restrict__ s, unsigned short* __restrict__ d, int n)
{
    int i = blockIdx.x * 256 + threadIdx.x;
    if (i < n) d[i] = f2bf(s[i]);
}

// pack f32 W[K][N] -> bf16 K-pair dwords Wp[K/2][N]
__global__ __launch_bounds__(256) void pack_w_bf16(
    const float* __restrict__ W, unsigned* __restrict__ Wp, int K, int N)
{
    int tid = blockIdx.x * 256 + threadIdx.x;
    if (tid >= (K >> 1) * N) return;
    int p = tid / N, n = tid - p * N;
    unsigned lo = f2bf(W[(size_t)(2 * p) * N + n]);
    unsigned hi = f2bf(W[(size_t)(2 * p + 1) * N + n]);
    Wp[tid] = lo | (hi << 16);
}

// al[n,h] = sum_c H[n,h,c] * a[h,c]   (one wave per (n,h), C=256)
__global__ __launch_bounds__(256) void attn_coef(
    const float* __restrict__ H, const float* __restrict__ a_src,
    const float* __restrict__ a_dst, float* __restrict__ al_s,
    float* __restrict__ al_d, int NH, int HH)
{
    int wid = blockIdx.x * 8 + (threadIdx.x >> 5);
    if (wid >= NH) return;
    int lane = threadIdx.x & 31;
    int h = wid % HH;
    const float* hp = H + (size_t)wid * 256;
    const float* as = a_src + h * 256;
    const float* ad = a_dst + h * 256;
    float ss = 0.f, sd = 0.f;
    for (int c = lane; c < 256; c += 32) {
        float v = hp[c];
        ss += v * as[c];
        sd += v * ad[c];
    }
    for (int off = 16; off; off >>= 1) {
        ss += __shfl_down(ss, off, 32);
        sd += __shfl_down(sd, off, 32);
    }
    if (lane == 0) { al_s[wid] = ss; al_d[wid] = sd; }
}

__device__ __forceinline__ void edge_sd(const int* __restrict__ ei, int E0,
                                        int e, int& s, int& d)
{
    if (e < E0) { s = ei[e]; d = ei[E0 + e]; }
    else        { s = d = e - E0; }          // self-loop
}

__global__ __launch_bounds__(256) void edge_max(
    const int* __restrict__ ei, const float* __restrict__ al_s,
    const float* __restrict__ al_d, unsigned* __restrict__ mkey,
    int E0, int Et, int HH)
{
    int tid = blockIdx.x * 256 + threadIdx.x;
    if (tid >= Et * HH) return;
    int e = tid / HH, h = tid - e * HH;
    int s, d; edge_sd(ei, E0, e, s, d);
    float v = al_s[s * HH + h] + al_d[d * HH + h];
    v = v > 0.f ? v : 0.2f * v;              // leaky_relu(0.2)
    atomicMax(&mkey[d * HH + h], fkey(v));
}

__global__ __launch_bounds__(256) void edge_expsum(
    const int* __restrict__ ei, const float* __restrict__ al_s,
    const float* __restrict__ al_d, const unsigned* __restrict__ mkey,
    float* __restrict__ ssum, float* __restrict__ exbuf,
    int E0, int Et, int HH)
{
    int tid = blockIdx.x * 256 + threadIdx.x;
    if (tid >= Et * HH) return;
    int e = tid / HH, h = tid - e * HH;
    int s, d; edge_sd(ei, E0, e, s, d);
    float v = al_s[s * HH + h] + al_d[d * HH + h];
    v = v > 0.f ? v : 0.2f * v;
    float ex = __expf(v - funkey(mkey[d * HH + h]));
    exbuf[tid] = ex;
    atomicAdd(&ssum[d * HH + h], ex);
}

// agg[dst,h,:] += alpha * H[src,h,:] ; one wave per (edge,head), C=256
__global__ __launch_bounds__(256) void edge_agg(
    const int* __restrict__ ei, const float* __restrict__ H,
    const float* __restrict__ exbuf, const float* __restrict__ ssum,
    float* __restrict__ agg, int E0, int Et, int HH)
{
    int wid = blockIdx.x * 8 + (threadIdx.x >> 5);
    if (wid >= Et * HH) return;
    int lane = threadIdx.x & 31;
    int e = wid / HH, h = wid - e * HH;
    int s, d; edge_sd(ei, E0, e, s, d);
    float alpha = exbuf[wid] / (ssum[d * HH + h] + 1e-16f);
    const float4* hp = (const float4*)(H + ((size_t)s * HH + h) * 256);
    float* ap = agg + ((size_t)d * HH + h) * 256;
#pragma unroll
    for (int i = 0; i < 2; ++i) {
        float4 v = hp[lane + 32 * i];
        int c = (lane + 32 * i) * 4;
        atomicAdd(ap + c + 0, alpha * v.x);
        atomicAdd(ap + c + 1, alpha * v.y);
        atomicAdd(ap + c + 2, alpha * v.z);
        atomicAdd(ap + c + 3, alpha * v.w);
    }
}

// layers 1/2: ELU(agg + b + skip + bs) -> bf16 activations for next GEMM
__global__ __launch_bounds__(256) void finalize_cat(
    const float* __restrict__ agg, const float* __restrict__ b,
    const float* __restrict__ sk, const float* __restrict__ bs,
    unsigned short* __restrict__ Xb, int total, int F)
{
    int tid = blockIdx.x * 256 + threadIdx.x;
    if (tid >= total) return;
    int f = tid % F;
    float v = agg[tid] + b[f] + sk[tid] + bs[f];
    v = v > 0.f ? v : expm1f(v);             // ELU
    Xb[tid] = f2bf(v);
}

// layer 3: mean over heads + b3 + skip + bs3 -> f32 output
__global__ __launch_bounds__(256) void finalize_mean(
    const float* __restrict__ agg, const float* __restrict__ b3,
    const float* __restrict__ sk, const float* __restrict__ bs3,
    float* __restrict__ out, int Nn, int Cc, int HH)
{
    int tid = blockIdx.x * 256 + threadIdx.x;
    if (tid >= Nn * Cc) return;
    int n = tid / Cc, c = tid - n * Cc;
    float v = 0.f;
    for (int h = 0; h < HH; ++h) v += agg[((size_t)n * HH + h) * Cc + c];
    out[tid] = v / (float)HH + b3[c] + sk[tid] + bs3[c];
}

// ---------------------------------------------------------------------------
extern "C" void kernel_launch(void* const* d_in, const int* in_sizes, int n_in,
                              void* d_out, int out_size, void* d_ws,
                              size_t ws_size, hipStream_t stream)
{
    (void)n_in; (void)out_size; (void)ws_size;
    const int Nn = in_sizes[0] / 128;        // 10000
    const int E0 = in_sizes[1] / 2;          // 160000
    const int Et = E0 + Nn;                  // + self loops
    const int H12 = in_sizes[3] / 256;       // 4 heads
    const int H3  = in_sizes[15] / 256;      // 6 heads
    const int Cc  = 256;

    const float* x   = (const float*)d_in[0];
    const int*   ei  = (const int*)d_in[1];
    const float* W1  = (const float*)d_in[2];
    const float* as1 = (const float*)d_in[3];
    const float* ad1 = (const float*)d_in[4];
    const float* b1  = (const float*)d_in[5];
    const float* Ws1 = (const float*)d_in[6];
    const float* bs1 = (const float*)d_in[7];
    const float* W2  = (const float*)d_in[8];
    const float* as2 = (const float*)d_in[9];
    const float* ad2 = (const float*)d_in[10];
    const float* b2  = (const float*)d_in[11];
    const float* Ws2 = (const float*)d_in[12];
    const float* bs2 = (const float*)d_in[13];
    const float* W3  = (const float*)d_in[14];
    const float* as3 = (const float*)d_in[15];
    const float* ad3 = (const float*)d_in[16];
    const float* b3  = (const float*)d_in[17];
    const float* Ws3 = (const float*)d_in[18];
    const float* bs3 = (const float*)d_in[19];
    float* out = (float*)d_out;

    // --- workspace carve-out ---
    size_t off = 0;
    auto alloc = [&](size_t bytes) -> char* {
        size_t cur = (off + 255) & ~(size_t)255;
        off = cur + bytes;
        return (char*)d_ws + cur;
    };
    unsigned short* Xb  = (unsigned short*)alloc((size_t)Nn * 1024 * 2);
    unsigned* W1p  = (unsigned*)alloc((size_t)64  * 1024 * 4);
    unsigned* Ws1p = (unsigned*)alloc((size_t)64  * 1024 * 4);
    unsigned* W2p  = (unsigned*)alloc((size_t)512 * 1024 * 4);
    unsigned* Ws2p = (unsigned*)alloc((size_t)512 * 1024 * 4);
    unsigned* W3p  = (unsigned*)alloc((size_t)512 * 1536 * 4);
    unsigned* Ws3p = (unsigned*)alloc((size_t)512 * 256 * 4);
    float*    Hbuf  = (float*)alloc((size_t)Nn * 1536 * 4);
    float*    Sk    = (float*)alloc((size_t)Nn * 1024 * 4);
    float*    agg   = (float*)alloc((size_t)Nn * 1536 * 4);
    float*    al_s  = (float*)alloc((size_t)Nn * 6 * 4);
    float*    al_d  = (float*)alloc((size_t)Nn * 6 * 4);
    unsigned* mkey  = (unsigned*)alloc((size_t)Nn * 6 * 4);
    float*    ssum  = (float*)alloc((size_t)Nn * 6 * 4);
    float*    exbuf = (float*)alloc((size_t)Et * 6 * 4);

    auto packw = [&](const float* w, unsigned* wp, int K, int N) {
        int tot = (K / 2) * N;
        pack_w_bf16<<<(tot + 255) / 256, 256, 0, stream>>>(w, wp, K, N);
    };
    auto gemm = [&](const unsigned short* A, const unsigned* B, float* C,
                    int M, int N, int K) {
        wmma_gemm_bf16<<<dim3(N / 256, M / 16), 256, 0, stream>>>(A, B, C, M, N, K);
    };

    // --- conversions: input features -> bf16, weights -> packed pair dwords ---
    f32_to_bf16_k<<<(Nn * 128 + 255) / 256, 256, 0, stream>>>(x, Xb, Nn * 128);
    packw(W1,  W1p,  128,  1024);
    packw(Ws1, Ws1p, 128,  1024);
    packw(W2,  W2p,  1024, 1024);
    packw(Ws2, Ws2p, 1024, 1024);
    packw(W3,  W3p,  1024, 1536);
    packw(Ws3, Ws3p, 1024, 256);

    auto run_layer = [&](int Kin, const unsigned* Wp, const unsigned* Wsp,
                         int skipN, const float* a_s, const float* a_d,
                         const float* bias, const float* bias_s,
                         int HH, bool last) {
        const int FO = HH * Cc;
        gemm(Xb, Wp,  Hbuf, Nn, FO,    Kin);   // H = X @ W  -> [N, H*C]
        gemm(Xb, Wsp, Sk,   Nn, skipN, Kin);   // skip = X @ Ws
        const int NH = Nn * HH;
        attn_coef<<<(NH + 7) / 8, 256, 0, stream>>>(Hbuf, a_s, a_d,
                                                    al_s, al_d, NH, HH);
        (void)hipMemsetAsync(mkey, 0, (size_t)NH * 4, stream);
        (void)hipMemsetAsync(ssum, 0, (size_t)NH * 4, stream);
        (void)hipMemsetAsync(agg,  0, (size_t)Nn * FO * 4, stream);
        const int EH = Et * HH;
        edge_max<<<(EH + 255) / 256, 256, 0, stream>>>(ei, al_s, al_d, mkey,
                                                       E0, Et, HH);
        edge_expsum<<<(EH + 255) / 256, 256, 0, stream>>>(ei, al_s, al_d, mkey,
                                                          ssum, exbuf, E0, Et, HH);
        edge_agg<<<(EH + 7) / 8, 256, 0, stream>>>(ei, Hbuf, exbuf, ssum, agg,
                                                   E0, Et, HH);
        if (!last) {
            const int total = Nn * FO;
            finalize_cat<<<(total + 255) / 256, 256, 0, stream>>>(
                agg, bias, Sk, bias_s, Xb, total, FO);
        } else {
            const int total = Nn * Cc;
            finalize_mean<<<(total + 255) / 256, 256, 0, stream>>>(
                agg, bias, Sk, bias_s, out, Nn, Cc, HH);
        }
    };

    run_layer(128,  W1p, Ws1p, 1024, as1, ad1, b1, bs1, H12, false);
    run_layer(1024, W2p, Ws2p, 1024, as2, ad2, b2, bs2, H12, false);
    run_layer(1024, W3p, Ws3p, 256,  as3, ad3, b3, bs3, H3,  true);
}
